// ESRNN_29618094474262
// MI455X (gfx1250) — matrix operand order repaired and checked by reading, not compiled, not run on hardware
//
#include <hip/hip_runtime.h>
#include <cmath>

// ---------------------------------------------------------------------------
// ES-RNN forward for MI455X (gfx1250, wave32, WMMA).
// Strategy:
//   - es_kernel: per-series exponential smoothing scan (B=128 threads).
//   - Input projections x@Wih^T + b hoisted into big f16 WMMA GEMMs.
//   - Per-scan-step kernels compute h@Whh^T via v_wmma_f32_16x16x32_f16;
//     each wave owns one 16x16 (row, hidden) tile and all 4 gate column
//     offsets, so the LSTM cell nonlinearity is computed in-register.
//   - Dilation padding: zero A rows => projection GEMM yields bias there,
//     matching the reference's zero-padded xt.
// Workspace use: ~172 MB.
// ---------------------------------------------------------------------------

typedef _Float16 f16;
typedef _Float16 half8  __attribute__((ext_vector_type(8)));
typedef _Float16 half16 __attribute__((ext_vector_type(16)));
typedef float    float8 __attribute__((ext_vector_type(8)));

#define B_    128
#define T_    300
#define L_    28
#define OUT_  28
#define E_    20
#define H_    256
#define W_    273          // T - L + 1
#define WOUT_ 245          // W - OUT
#define FPAD  64           // F_IN=49 padded to 64 for K%32==0
#define SEXT  314          // S_ext columns

#define MROWS   (W_ * B_)        // 34944
#define MROWS2  (274 * B_)       // 35072  (dilation-2 padded)
#define MROWS4  (276 * B_)       // 35328  (dilation-6 padded)

// d_out layout (floats): pred | actual | holdout | rnn_out | hav | hav_norm
#define O_PRED  0
#define O_ACT   (WOUT_ * B_ * OUT_)                 // 878080
#define O_HOLD  (O_ACT  + WOUT_ * B_ * OUT_)        // 1756160
#define O_RNN   (O_HOLD + B_ * OUT_)                // 1759744
#define O_HAV   (O_RNN  + W_ * B_ * OUT_)           // 2738176
#define O_HAVN  (O_HAV  + B_ * OUT_)                // 2741760

// ---------------- fragment helpers (CDNA5 16x16x32 f16 layout) -------------

__device__ __forceinline__ half16 mk_afrag(const f16* __restrict__ row, int kk, int hsel) {
    // A 16x32: lane = M row; half-wave selects contiguous K block (0..15 / 16..31).
    return *reinterpret_cast<const half16*>(row + kk + hsel * 16);
}

__device__ __forceinline__ float sigm(float x) { return 1.0f / (1.0f + __expf(-x)); }

// ---------------- exponential smoothing scan -------------------------------

__global__ void es_kernel(const float* __restrict__ x,
                          const float* __restrict__ alpha_raw,
                          const float* __restrict__ gamma_raw,
                          const float* __restrict__ init_seas,
                          float* __restrict__ levels,      // B x T
                          float* __restrict__ Sext) {      // B x 314
    int b = threadIdx.x;
    if (b >= B_) return;
    float a = sigm(alpha_raw[b]);
    float g = sigm(gamma_raw[b]);
    const float* xr = x + b * T_;
    float* S  = Sext + b * SEXT;
    float* lv = levels + b * T_;
    float s00 = 0.0f;
#pragma unroll
    for (int j = 0; j < 7; ++j) {
        float s0 = __expf(init_seas[b * 7 + j]);
        S[j] = s0;
        if (j == 0) s00 = s0;
    }
    S[7] = s00;                       // s7
    float level = xr[0] / s00;
    lv[0] = level;
    for (int t = 1; t < T_; ++t) {
        float s   = S[t];
        float lvl = a * (xr[t] / s) + (1.0f - a) * level;
        float sn  = g * (xr[t] / lvl) + (1.0f - g) * s;
        S[t + 7] = sn;
        lv[t]    = lvl;
        level    = lvl;
    }
#pragma unroll
    for (int j = 0; j < 7; ++j) S[T_ + 7 + j] = S[T_ + j];   // S_ext tail
}

// ---------------- elementwise builders -------------------------------------

__global__ void build_winin_kernel(const float* __restrict__ x,
                                   const float* __restrict__ Sext,
                                   const float* __restrict__ levels,
                                   const float* __restrict__ cats,
                                   const float* __restrict__ my_param,
                                   f16* __restrict__ winin) {      // (W*B) x 64
    int i = blockIdx.x * blockDim.x + threadIdx.x;
    if (i >= MROWS * FPAD) return;
    int f  = i & (FPAD - 1);
    int wb = i >> 6;
    int b  = wb % B_;
    int w  = wb / B_;
    int bb = b % 10;                   // reference's [arange(B) % 10] gather
    float v;
    if (f < L_)            v = x[bb * T_ + w + f] / Sext[bb * SEXT + w + f]
                               / levels[bb * T_ + (L_ - 1) + w];
    else if (f < L_ + E_)  v = cats[bb * E_ + (f - L_)];
    else if (f == L_ + E_) v = my_param[0];
    else                   v = 0.0f;   // K padding
    winin[i] = (f16)v;
}

__global__ void build_actual_kernel(const float* __restrict__ x,
                                    const float* __restrict__ Sext,
                                    const float* __restrict__ levels,
                                    float* __restrict__ out) {     // at O_ACT
    int i = blockIdx.x * blockDim.x + threadIdx.x;
    if (i >= WOUT_ * B_ * OUT_) return;
    int j  = i % OUT_;
    int wb = i / OUT_;
    int b  = wb % B_;
    int w  = wb / B_;
    out[i] = x[b * T_ + L_ + w + j] / Sext[b * SEXT + L_ + w + j]
             / levels[b * T_ + (L_ - 1) + w];
}

__global__ void convert_pad_kernel(const float* __restrict__ src, f16* __restrict__ dst,
                                   int srows, int scols, int dcols, int total) {
    int i = blockIdx.x * blockDim.x + threadIdx.x;
    if (i >= total) return;
    int r = i / dcols, c = i % dcols;
    float v = (r < srows && c < scols) ? src[r * scols + c] : 0.0f;
    dst[i] = (f16)v;
}

__global__ void zero_f16_kernel(f16* __restrict__ p, int n) {
    int i = blockIdx.x * blockDim.x + threadIdx.x;
    if (i < n) p[i] = (f16)0.0f;
}

__global__ void add_f16_kernel(f16* __restrict__ dst, const f16* __restrict__ src, int n) {
    int i = blockIdx.x * blockDim.x + threadIdx.x;
    if (i < n) dst[i] = (f16)((float)dst[i] + (float)src[i]);
}

// ---------------- generic WMMA GEMM: C = A(f16,MxK) * Wt(f16,NxK)^T + bias --
// One wave per 16x16 tile. M multiple of 16 guaranteed by caller.

template <int ACT, int OUT_F16>
__global__ void wmma_gemm_kernel(const f16* __restrict__ A, int lda,
                                 const f16* __restrict__ Wt, int ldw,
                                 const float* __restrict__ bias,
                                 void* __restrict__ Cout, int ldc,
                                 int Nreal, int K) {
    const int lane = threadIdx.x;
    const int hsel = lane >> 4;
    const int mr   = lane & 15;
    const int m0   = blockIdx.x * 16;
    const int n0   = blockIdx.y * 16;
    const f16* arow = A  + (size_t)(m0 + mr) * lda;
    const f16* brow = Wt + (size_t)(n0 + mr) * ldw;
    float8 acc = {};
    for (int kk = 0; kk < K; kk += 32) {
        half16 af = mk_afrag(arow, kk, hsel);
        half16 bf = *reinterpret_cast<const half16*>(brow + kk + hsel * 16);
        acc = __builtin_amdgcn_wmma_f32_16x16x32_f16(false, af, false, bf,
                                                     (short)0, acc, false, false);
    }
    const int col = n0 + mr;
    if (col >= Nreal) return;                 // after all WMMA (EXEC all-1 during MMA)
    const float bv = bias ? bias[col] : 0.0f;
#pragma unroll
    for (int r = 0; r < 8; ++r) {
        const int row = m0 + r + hsel * 8;
        float v = acc[r] + bv;
        if (ACT == 1) v = tanhf(v);
        if (OUT_F16)
            reinterpret_cast<f16*>(Cout)[(size_t)row * ldc + col] = (f16)v;
        else
            reinterpret_cast<float*>(Cout)[(size_t)row * ldc + col] = v;
    }
}

// ---------------- LSTM recurrent step ---------------------------------------
// One wave computes a 16x16 (row, hidden) tile for ALL 4 gates (column
// offsets 0/256/512/768 in z), so the cell update is fully in-register.
// grid = (dB/16, 16), block = 32 (one wave).

__global__ void lstm_step_kernel(const f16* __restrict__ Xp,     // Mpad x 1024 (incl. bias)
                                 f16* __restrict__ Hbuf,         // layer out (rows x 256)
                                 float* __restrict__ Cbuf,       // dB x 256
                                 const f16* __restrict__ Whh,    // 1024 x 256
                                 int row0, int prev_row0) {
    const int lane = threadIdx.x;
    const int hsel = lane >> 4;
    const int mr   = lane & 15;
    const int m0   = blockIdx.x * 16;    // local row tile
    const int c0   = blockIdx.y * 16;    // hidden column tile
    float8 acc[4];
#pragma unroll
    for (int g = 0; g < 4; ++g) {
        const int col = g * H_ + c0 + mr;
#pragma unroll
        for (int r = 0; r < 8; ++r) {
            const int row = row0 + m0 + r + hsel * 8;
            acc[g][r] = (float)Xp[(size_t)row * (4 * H_) + col];
        }
    }
    if (prev_row0 >= 0) {   // scalar-uniform branch; h_prev==0 at first step
        const f16* arow = Hbuf + (size_t)(prev_row0 + m0 + mr) * H_;
        for (int kk = 0; kk < H_; kk += 32) {
            half16 af = mk_afrag(arow, kk, hsel);
#pragma unroll
            for (int g = 0; g < 4; ++g) {
                const f16* brow = Whh + (size_t)(g * H_ + c0 + mr) * H_ + kk + hsel * 16;
                half16 bf = *reinterpret_cast<const half16*>(brow);
                acc[g] = __builtin_amdgcn_wmma_f32_16x16x32_f16(false, af, false, bf,
                                                                (short)0, acc[g], false, false);
            }
        }
    }
#pragma unroll
    for (int r = 0; r < 8; ++r) {
        const int lrow = m0 + r + hsel * 8;
        const int hcol = c0 + mr;
        const float zi = acc[0][r], zf = acc[1][r], zg = acc[2][r], zo = acc[3][r];
        const float cp = (prev_row0 >= 0) ? Cbuf[(size_t)lrow * H_ + hcol] : 0.0f;
        const float cn = sigm(zf) * cp + sigm(zi) * tanhf(zg);
        const float hv = sigm(zo) * tanhf(cn);
        Cbuf[(size_t)lrow * H_ + hcol] = cn;
        Hbuf[(size_t)(row0 + lrow) * H_ + hcol] = (f16)hv;
    }
}

// ---------------- output assembly -------------------------------------------

__global__ void copy_pred_kernel(float* __restrict__ out) {
    int i = blockIdx.x * blockDim.x + threadIdx.x;
    if (i >= WOUT_ * B_ * OUT_) return;
    out[O_PRED + i] = out[O_RNN + i];
}

__global__ void tail_kernel(const float* __restrict__ val,
                            const float* __restrict__ Sext,
                            const float* __restrict__ levels,
                            float* __restrict__ out) {
    int i = blockIdx.x * blockDim.x + threadIdx.x;
    if (i >= B_ * OUT_) return;
    int b = i / OUT_, j = i % OUT_;
    float se = Sext[b * SEXT + (SEXT - OUT_) + j];
    float lv = levels[b * T_ + (T_ - 1)];
    float r  = out[O_RNN + ((W_ - 1) * B_ + b) * OUT_ + j];
    float hv = r * se * lv;
    out[O_HOLD + i] = hv > 0.0f ? hv : 0.0f;
    float vv = val[i];
    out[O_HAV  + i] = vv;
    out[O_HAVN + i] = vv / (se * lv);
}

// ---------------- host orchestration ----------------------------------------

static inline dim3 g1(int n) { return dim3((n + 255) / 256); }

extern "C" void kernel_launch(void* const* d_in, const int* in_sizes, int n_in,
                              void* d_out, int out_size, void* d_ws, size_t ws_size,
                              hipStream_t stream) {
    (void)in_sizes; (void)n_in; (void)out_size; (void)ws_size;
    const float* x      = (const float*)d_in[0];
    const float* val    = (const float*)d_in[1];
    const float* araw   = (const float*)d_in[2];
    const float* graw   = (const float*)d_in[3];
    const float* iseas  = (const float*)d_in[4];
    const float* cats   = (const float*)d_in[5];
    const float* mparam = (const float*)d_in[6];
    const float* Wih[4] = {(const float*)d_in[7],  (const float*)d_in[10],
                           (const float*)d_in[13], (const float*)d_in[16]};
    const float* Whh[4] = {(const float*)d_in[8],  (const float*)d_in[11],
                           (const float*)d_in[14], (const float*)d_in[17]};
    const float* bb[4]  = {(const float*)d_in[9],  (const float*)d_in[12],
                           (const float*)d_in[15], (const float*)d_in[18]};
    const float* linW = (const float*)d_in[19];
    const float* linB = (const float*)d_in[20];
    const float* scW  = (const float*)d_in[21];
    const float* scB  = (const float*)d_in[22];
    float* out = (float*)d_out;

    // ---- workspace carve-up (256B aligned) ----
    char* p = (char*)d_ws;
    auto carve = [&](size_t bytes) -> void* {
        void* r = p; p += (bytes + 255) & ~(size_t)255; return r;
    };
    float* levels = (float*)carve((size_t)B_ * T_ * 4);
    float* Sext   = (float*)carve((size_t)B_ * SEXT * 4);
    f16* winin    = (f16*)carve((size_t)MROWS * FPAD * 2);
    f16* Wih1h    = (f16*)carve((size_t)4 * H_ * FPAD * 2);
    f16* WihH[4]; WihH[0] = Wih1h;
    for (int l = 1; l < 4; ++l) WihH[l] = (f16*)carve((size_t)4 * H_ * H_ * 2);
    f16* WhhH[4];
    for (int l = 0; l < 4; ++l) WhhH[l] = (f16*)carve((size_t)4 * H_ * H_ * 2);
    f16* linWh   = (f16*)carve((size_t)H_ * H_ * 2);
    f16* scWh    = (f16*)carve((size_t)32 * H_ * 2);
    f16* Xp      = (f16*)carve((size_t)MROWS4 * 4 * H_ * 2);
    f16* H1      = (f16*)carve((size_t)MROWS4 * H_ * 2);
    f16* H1b     = (f16*)carve((size_t)MROWS4 * H_ * 2);
    f16* H3      = (f16*)carve((size_t)MROWS4 * H_ * 2);
    f16* H4      = (f16*)carve((size_t)MROWS4 * H_ * 2);
    float* Cbuf  = (float*)carve((size_t)768 * H_ * 4);
    f16* hfin    = (f16*)carve((size_t)MROWS * H_ * 2);

    // ---- phase 0: ES scan + derived tensors ----
    es_kernel<<<1, B_, 0, stream>>>(x, araw, graw, iseas, levels, Sext);
    build_winin_kernel<<<g1(MROWS * FPAD), 256, 0, stream>>>(x, Sext, levels, cats, mparam, winin);
    build_actual_kernel<<<g1(WOUT_ * B_ * OUT_), 256, 0, stream>>>(x, Sext, levels, out + O_ACT);

    // ---- weight conversion (f32 -> f16, K padded where needed) ----
    convert_pad_kernel<<<g1(4 * H_ * FPAD), 256, 0, stream>>>(Wih[0], Wih1h, 4 * H_, L_ + 1 + E_, FPAD, 4 * H_ * FPAD);
    for (int l = 1; l < 4; ++l)
        convert_pad_kernel<<<g1(4 * H_ * H_), 256, 0, stream>>>(Wih[l], WihH[l], 4 * H_, H_, H_, 4 * H_ * H_);
    for (int l = 0; l < 4; ++l)
        convert_pad_kernel<<<g1(4 * H_ * H_), 256, 0, stream>>>(Whh[l], WhhH[l], 4 * H_, H_, H_, 4 * H_ * H_);
    convert_pad_kernel<<<g1(H_ * H_), 256, 0, stream>>>(linW, linWh, H_, H_, H_, H_ * H_);
    convert_pad_kernel<<<g1(32 * H_), 256, 0, stream>>>(scW, scWh, OUT_, H_, H_, 32 * H_);

    // ---- zero H buffers (pad rows must be 0 when consumed as GEMM A) ----
    const int hN = MROWS4 * H_;
    zero_f16_kernel<<<g1(hN), 256, 0, stream>>>(H1,  hN);
    zero_f16_kernel<<<g1(hN), 256, 0, stream>>>(H1b, hN);
    zero_f16_kernel<<<g1(hN), 256, 0, stream>>>(H3,  hN);
    zero_f16_kernel<<<g1(hN), 256, 0, stream>>>(H4,  hN);

    // ---- layer 1: d=1 ----
    wmma_gemm_kernel<0, 1><<<dim3(MROWS / 16, 64), 32, 0, stream>>>(
        winin, FPAD, Wih1h, FPAD, bb[0], Xp, 4 * H_, 4 * H_, FPAD);
    for (int n = 0; n < W_; ++n)
        lstm_step_kernel<<<dim3(B_ / 16, 16), 32, 0, stream>>>(
            Xp, H1, Cbuf, WhhH[0], n * B_, n == 0 ? -1 : (n - 1) * B_);

    // ---- layer 2: d=2 (Tpad=274) ----
    wmma_gemm_kernel<0, 1><<<dim3(MROWS2 / 16, 64), 32, 0, stream>>>(
        H1, H_, WihH[1], H_, bb[1], Xp, 4 * H_, 4 * H_, H_);
    for (int n = 0; n < 137; ++n)
        lstm_step_kernel<<<dim3(2 * B_ / 16, 16), 32, 0, stream>>>(
            Xp, H1b, Cbuf, WhhH[1], n * 2 * B_, n == 0 ? -1 : (n - 1) * 2 * B_);
    // trim: zero padded-time block [273B, 274B) before it feeds layer 3
    zero_f16_kernel<<<g1(B_ * H_), 256, 0, stream>>>(H1b + (size_t)W_ * B_ * H_, B_ * H_);

    // ---- layer 3: d=2 ----
    wmma_gemm_kernel<0, 1><<<dim3(MROWS2 / 16, 64), 32, 0, stream>>>(
        H1b, H_, WihH[2], H_, bb[2], Xp, 4 * H_, 4 * H_, H_);
    for (int n = 0; n < 137; ++n)
        lstm_step_kernel<<<dim3(2 * B_ / 16, 16), 32, 0, stream>>>(
            Xp, H3, Cbuf, WhhH[2], n * 2 * B_, n == 0 ? -1 : (n - 1) * 2 * B_);
    zero_f16_kernel<<<g1(B_ * H_), 256, 0, stream>>>(H3 + (size_t)W_ * B_ * H_, B_ * H_);

    // ---- layer 4: d=6 (Tpad=276) ----
    wmma_gemm_kernel<0, 1><<<dim3(MROWS4 / 16, 64), 32, 0, stream>>>(
        H3, H_, WihH[3], H_, bb[3], Xp, 4 * H_, 4 * H_, H_);
    for (int n = 0; n < 46; ++n)
        lstm_step_kernel<<<dim3(6 * B_ / 16, 16), 32, 0, stream>>>(
            Xp, H4, Cbuf, WhhH[3], n * 6 * B_, n == 0 ? -1 : (n - 1) * 6 * B_);

    // ---- residual + head ----
    add_f16_kernel<<<g1(MROWS * H_), 256, 0, stream>>>(H4, H1b, MROWS * H_);
    wmma_gemm_kernel<1, 1><<<dim3(MROWS / 16, H_ / 16), 32, 0, stream>>>(
        H4, H_, linWh, H_, linB, hfin, H_, H_, H_);
    wmma_gemm_kernel<0, 0><<<dim3(MROWS / 16, 2), 32, 0, stream>>>(
        hfin, H_, scWh, H_, scB, out + O_RNN, OUT_, OUT_, H_);

    // ---- output assembly ----
    copy_pred_kernel<<<g1(WOUT_ * B_ * OUT_), 256, 0, stream>>>(out);
    tail_kernel<<<g1(B_ * OUT_), 256, 0, stream>>>(val, Sext, levels, out);
}
